// GlobalGraphBranch_88330297409788
// MI455X (gfx1250) — matrix operand order repaired and checked by reading, not compiled, not run on hardware
//
#include <hip/hip_runtime.h>

// ---------------------------------------------------------------------------
// GNN layer: x = feat @ Wp + bp ; msgs = x[src]*w ; agg = segsum(msgs, dst)
//            out = relu(agg @ Wa + ba)
// MI455X (gfx1250) strategy:
//   * Both GEMMs via v_wmma_f32_16x16x32_bf16 (bf16 inputs, f32 accum).
//     f32->bf16 via native converts (clang lowers fptrunc to v_cvt_*bf16*).
//   * Scatter via global_atomic_add_f32 — agg (25.6 MB) is L2-resident.
//   * Whole problem is HBM-bound (~100 MB => ~4.5us @ 23.3 TB/s), so bf16
//     A/B costs nothing in time and uses the native CDNA5 matrix path.
// ---------------------------------------------------------------------------

typedef __attribute__((ext_vector_type(16))) __bf16 v16bf;
typedef __attribute__((ext_vector_type(8)))  float  v8f;

// C = A[M,KDIM] * W[KDIM,128] + bias, optional ReLU.
// Block: 256 threads = 8 waves; wave w -> 16x16 tile at cols [16w,16w+16).
// Grid.x: M/16 tiles. WMMA: 16x16x32 bf16 -> f32.
template <int KDIM, bool RELU>
__global__ __launch_bounds__(256)
void wmma_gemm_128(const float* __restrict__ A,
                   const float* __restrict__ W,
                   const float* __restrict__ bias,
                   float* __restrict__ out,
                   int M)
{
  const int lane  = threadIdx.x & 31;
  const int wave  = threadIdx.x >> 5;
  const int tileM = blockIdx.x << 4;
  const int tileN = wave << 4;

  // A-fragment addressing per ISA 16-bit A 16x32 layout:
  // lanes 0-15: K = [0..7] U [16..23]; lanes 16-31: K = [8..15] U [24..31]
  int arow = tileM + (lane & 15);
  if (arow >= M) arow = M - 1;        // clamp: keep EXEC all-ones for WMMA
  const int kb = (lane < 16) ? 0 : 8;

  v8f c = {};
  #pragma unroll
  for (int k0 = 0; k0 < KDIM; k0 += 32) {
    // ---- A fragment: 16 bf16 per lane (two 8-float contiguous chunks) ----
    const float* ap = A + (size_t)arow * KDIM + k0 + kb;
    const float4 a0 = *(const float4*)(ap +  0);
    const float4 a1 = *(const float4*)(ap +  4);
    const float4 a2 = *(const float4*)(ap + 16);
    const float4 a3 = *(const float4*)(ap + 20);

    // ---- B fragment: lane L holds col tileN+L%16, K = kb+[0..7] U kb+[16..23] ----
    const float* bp = W + (size_t)(k0 + kb) * 128 + tileN + (lane & 15);
    const float4 b0 = make_float4(bp[0 * 128], bp[1 * 128], bp[2 * 128], bp[3 * 128]);
    const float4 b1 = make_float4(bp[4 * 128], bp[5 * 128], bp[6 * 128], bp[7 * 128]);
    const float4 b2 = make_float4(bp[16 * 128], bp[17 * 128], bp[18 * 128], bp[19 * 128]);
    const float4 b3 = make_float4(bp[20 * 128], bp[21 * 128], bp[22 * 128], bp[23 * 128]);

    // Native f32->bf16 converts (v_cvt_pk_bf16_f32 class on gfx1250).
    v16bf av, bv;
    av[0]  = (__bf16)a0.x; av[1]  = (__bf16)a0.y;
    av[2]  = (__bf16)a0.z; av[3]  = (__bf16)a0.w;
    av[4]  = (__bf16)a1.x; av[5]  = (__bf16)a1.y;
    av[6]  = (__bf16)a1.z; av[7]  = (__bf16)a1.w;
    av[8]  = (__bf16)a2.x; av[9]  = (__bf16)a2.y;
    av[10] = (__bf16)a2.z; av[11] = (__bf16)a2.w;
    av[12] = (__bf16)a3.x; av[13] = (__bf16)a3.y;
    av[14] = (__bf16)a3.z; av[15] = (__bf16)a3.w;

    bv[0]  = (__bf16)b0.x; bv[1]  = (__bf16)b0.y;
    bv[2]  = (__bf16)b0.z; bv[3]  = (__bf16)b0.w;
    bv[4]  = (__bf16)b1.x; bv[5]  = (__bf16)b1.y;
    bv[6]  = (__bf16)b1.z; bv[7]  = (__bf16)b1.w;
    bv[8]  = (__bf16)b2.x; bv[9]  = (__bf16)b2.y;
    bv[10] = (__bf16)b2.z; bv[11] = (__bf16)b2.w;
    bv[12] = (__bf16)b3.x; bv[13] = (__bf16)b3.y;
    bv[14] = (__bf16)b3.z; bv[15] = (__bf16)b3.w;

    // (neg_a, A, neg_b, B, c_mod, C, reuse_a, reuse_b)
    c = __builtin_amdgcn_wmma_f32_16x16x32_bf16(false, av, false, bv,
                                                (short)0, c, false, false);
  }

  // C/D layout: elem j -> row tileM + j + (lane<16?0:8), col tileN + lane%16
  const int   col = tileN + (lane & 15);
  const float bb  = bias[col];
  const int   r0  = tileM + ((lane < 16) ? 0 : 8);
  #pragma unroll
  for (int j = 0; j < 8; ++j) {
    const int r = r0 + j;
    if (r < M) {
      float v = c[j] + bb;
      if (RELU) v = fmaxf(v, 0.0f);
      out[(size_t)r * 128 + col] = v;
    }
  }
}

// Zero agg (float4-vectorized).
__global__ __launch_bounds__(256)
void zero_f32_vec(float4* __restrict__ p, size_t n4) {
  size_t i = (size_t)blockIdx.x * blockDim.x + threadIdx.x;
  if (i < n4) p[i] = float4{0.f, 0.f, 0.f, 0.f};
}

// One wave per edge: lane covers 4 of the 128 features.
// Gathers x[src] (L2-resident) and scatter-adds into agg[dst] with f32 atomics.
__global__ __launch_bounds__(256)
void scatter_edges(const float* __restrict__ x,
                   const long long* __restrict__ edge_index, // [2,E] int64
                   const float* __restrict__ edge_weight,    // [E]
                   float* __restrict__ agg,
                   int E)
{
  const int e = blockIdx.x * 8 + (threadIdx.x >> 5);
  if (e >= E) return;
  const int lane = threadIdx.x & 31;

  const long long src = edge_index[e];             // row 0
  const long long dst = edge_index[(size_t)E + e]; // row 1
  const float     w   = edge_weight[e];

  const float4 v = *(const float4*)(x + (size_t)src * 128 + lane * 4);
  float* ad = agg + (size_t)dst * 128 + lane * 4;
  atomicAdd(ad + 0, v.x * w);
  atomicAdd(ad + 1, v.y * w);
  atomicAdd(ad + 2, v.z * w);
  atomicAdd(ad + 3, v.w * w);
}

extern "C" void kernel_launch(void* const* d_in, const int* in_sizes, int n_in,
                              void* d_out, int out_size, void* d_ws, size_t ws_size,
                              hipStream_t stream) {
  const float*     features    = (const float*)d_in[0];
  const long long* edge_index  = (const long long*)d_in[1]; // int64 per reference
  const float*     edge_weight = (const float*)d_in[2];
  const float*     W_proj      = (const float*)d_in[3];
  const float*     b_proj      = (const float*)d_in[4];
  const float*     W_agg       = (const float*)d_in[5];
  const float*     b_agg       = (const float*)d_in[6];
  float*           out         = (float*)d_out;

  const int D_IN = 256, D_H = 128;
  const int Nn = in_sizes[0] / D_IN;   // 50000
  const int E  = in_sizes[2];          // 800000

  // Workspace: x[Nn,128] then agg[Nn,128] (f32) = 51.2 MB total.
  float* x   = (float*)d_ws;
  float* agg = x + (size_t)Nn * D_H;

  const int mtiles = (Nn + 15) / 16;   // 3125 (exact)

  // 1) x = feat @ W_proj + b_proj   (WMMA bf16 -> f32)
  wmma_gemm_128<256, false><<<mtiles, 256, 0, stream>>>(features, W_proj, b_proj, x, Nn);

  // 2) agg = 0
  const size_t n4 = (size_t)Nn * D_H / 4;
  zero_f32_vec<<<(unsigned)((n4 + 255) / 256), 256, 0, stream>>>((float4*)agg, n4);

  // 3) agg[dst] += x[src] * w      (f32 atomics, resolved in L2)
  scatter_edges<<<(E + 7) / 8, 256, 0, stream>>>(x, edge_index, edge_weight, agg, E);

  // 4) out = relu(agg @ W_agg + b_agg)   (WMMA bf16 -> f32)
  wmma_gemm_128<128, true><<<mtiles, 256, 0, stream>>>(agg, W_agg, b_agg, out, Nn);
}
